// MemoryEfficientVocabOutput_87445534147042
// MI455X (gfx1250) — compile-verified, matrix operand order checked by reading, not run on hardware
//
#include <hip/hip_runtime.h>
#include <hip/hip_bf16.h>
#include <math.h>

// Problem constants (match reference)
#define TOKENS 4096
#define DMODEL 2048
#define VOCAB  32000

// Tiling
#define TM 128                 // tokens per block
#define VN 128                 // vocab rows per block
#define KC 64                  // K chunk through LDS
#define LDT 72                 // padded LDS row stride in bf16 (KC + 8) -> conflict-free b128 frag loads
#define TTILES (TOKENS / TM)   // 32
#define VTILES (VOCAB / VN)    // 250

typedef __attribute__((ext_vector_type(16))) __bf16 v16bf;
typedef __attribute__((ext_vector_type(8)))  __bf16 v8bf;
typedef __attribute__((ext_vector_type(4)))  __bf16 v4bf;
typedef __attribute__((ext_vector_type(8)))  float  v8f;

// Load one 16x32 bf16 WMMA fragment from LDS (row-major, LDT stride).
// Per ISA: lanes 0-15 hold row M=l15, K 0..7 (VGPR0-3) and K 16..23 (VGPR4-7);
// lanes 16-31 hold K 8..15 and K 24..31. Caller bakes (row,half) into p.
__device__ __forceinline__ v16bf load_frag(const __bf16* p) {
  v8bf lo = *(const v8bf*)(p);        // K = half*8 + [0..7]
  v8bf hi = *(const v8bf*)(p + 16);   // K = half*8 + 16 + [0..7]
  return __builtin_shufflevector(lo, hi, 0,1,2,3,4,5,6,7,8,9,10,11,12,13,14,15);
}

// Stage a 128-row x 64-col f32 tile into registers (8 float4 per thread).
__device__ __forceinline__ void load_global_tile(float4 (&r)[8], const float* src) {
  const int t = threadIdx.x;
#pragma unroll
  for (int i = 0; i < 8; ++i) {
    int f   = t + i * 256;        // float4 index within tile (0..2047)
    int row = f >> 4;             // 16 float4 per 64-col row
    int c4  = f & 15;
    r[i] = *(const float4*)(src + (size_t)row * DMODEL + (c4 << 2));
  }
}

// Convert staged registers to bf16 and store into a padded LDS tile.
__device__ __forceinline__ void store_tile_bf16(__bf16* dst, const float4 (&r)[8]) {
  const int t = threadIdx.x;
#pragma unroll
  for (int i = 0; i < 8; ++i) {
    int f   = t + i * 256;
    int row = f >> 4;
    int c4  = f & 15;
    v4bf h = { (__bf16)r[i].x, (__bf16)r[i].y, (__bf16)r[i].z, (__bf16)r[i].w };
    *(v4bf*)(dst + row * LDT + (c4 << 2)) = h;      // ds_store_b64
  }
}

__global__ __launch_bounds__(256)
void ce_logits_partial(const float* __restrict__ x, const float* __restrict__ w,
                       const int* __restrict__ tgt,
                       float2* __restrict__ ws_ms, float* __restrict__ ws_tgt)
{
  __shared__ __align__(16) __bf16 ldsX[2][TM * LDT];
  __shared__ __align__(16) __bf16 ldsW[2][VN * LDT];
  __shared__ float2 mergeMS[2][TM];   // [n-wave][row] per-wave (max, sumexp)

  const int tbase = blockIdx.x * TM;      // token tile (fast dim -> w-slice reuse in L2)
  const int nbase = blockIdx.y * VN;      // vocab tile

  const int t     = threadIdx.x;
  const int lane  = t & 31;
  const int wv    = t >> 5;               // 8 waves
  const int mwave = wv & 3;               // rows  [mwave*32, +32)
  const int nwave = wv >> 2;              // cols  [nwave*64, +64)
  const int l15   = lane & 15;
  const int half  = lane >> 4;

  v8f acc[2][4] = {};                     // 2 M-subtiles x 4 N-subtiles of 16x16 f32

  const float* xg = x + (size_t)tbase * DMODEL;
  const float* wg = w + (size_t)nbase * DMODEL;

  // Prologue: fill buffer 0
  float4 rx[8], rw[8];
  load_global_tile(rx, xg);
  load_global_tile(rw, wg);
  store_tile_bf16(ldsX[0], rx);
  store_tile_bf16(ldsW[0], rw);

  int cur = 0;
  for (int k0 = 0; k0 < DMODEL; k0 += KC) {
    __syncthreads();                       // buffer `cur` fully written by all waves
    const bool has_next = (k0 + KC) < DMODEL;
    if (has_next) {                        // issue next chunk's global loads early;
      load_global_tile(rx, xg + k0 + KC);  // independent of compute below -> latency hidden
      load_global_tile(rw, wg + k0 + KC);
    }

    const __bf16* xb = ldsX[cur] + (mwave * 32 + l15) * LDT + half * 8;
    const __bf16* wb = ldsW[cur] + (nwave * 64 + l15) * LDT + half * 8;
#pragma unroll
    for (int kk = 0; kk < KC; kk += 32) {
      v16bf a0 = load_frag(xb + kk);
      v16bf a1 = load_frag(xb + 16 * LDT + kk);
      v16bf b0 = load_frag(wb + kk);
      v16bf b1 = load_frag(wb + 16 * LDT + kk);
      v16bf b2 = load_frag(wb + 32 * LDT + kk);
      v16bf b3 = load_frag(wb + 48 * LDT + kk);
      acc[0][0] = __builtin_amdgcn_wmma_f32_16x16x32_bf16(false, a0, false, b0, (short)0, acc[0][0], false, false);
      acc[0][1] = __builtin_amdgcn_wmma_f32_16x16x32_bf16(false, a0, false, b1, (short)0, acc[0][1], false, false);
      acc[0][2] = __builtin_amdgcn_wmma_f32_16x16x32_bf16(false, a0, false, b2, (short)0, acc[0][2], false, false);
      acc[0][3] = __builtin_amdgcn_wmma_f32_16x16x32_bf16(false, a0, false, b3, (short)0, acc[0][3], false, false);
      acc[1][0] = __builtin_amdgcn_wmma_f32_16x16x32_bf16(false, a1, false, b0, (short)0, acc[1][0], false, false);
      acc[1][1] = __builtin_amdgcn_wmma_f32_16x16x32_bf16(false, a1, false, b1, (short)0, acc[1][1], false, false);
      acc[1][2] = __builtin_amdgcn_wmma_f32_16x16x32_bf16(false, a1, false, b2, (short)0, acc[1][2], false, false);
      acc[1][3] = __builtin_amdgcn_wmma_f32_16x16x32_bf16(false, a1, false, b3, (short)0, acc[1][3], false, false);
    }

    if (has_next) {                        // convert + park next chunk in the other buffer
      store_tile_bf16(ldsX[cur ^ 1], rx);
      store_tile_bf16(ldsW[cur ^ 1], rw);
    }
    cur ^= 1;
  }

  // ---- Epilogue: target-logit extraction (unique writer per token) ----
  // C/D layout: VGPR r -> row = subtile + r + half*8, col = nsub*16 + l15.
#pragma unroll
  for (int m = 0; m < 2; ++m) {
#pragma unroll
    for (int r = 0; r < 8; ++r) {
      const int row = mwave * 32 + m * 16 + r + half * 8;
      const int g   = tgt[tbase + row] - nbase;
      if ((unsigned)g < (unsigned)VN) {
        const int cg = g - nwave * 64;
        if ((unsigned)cg < 64u && (cg & 15) == l15) {
#pragma unroll
          for (int n = 0; n < 4; ++n)
            if ((cg >> 4) == n) ws_tgt[tbase + row] = acc[m][n][r];
        }
      }
    }
  }

  // ---- Epilogue: per-row online-softmax partials over this vocab tile ----
#pragma unroll
  for (int m = 0; m < 2; ++m) {
#pragma unroll
    for (int r = 0; r < 8; ++r) {
      float v = -3.402823466e38f;
#pragma unroll
      for (int n = 0; n < 4; ++n) v = fmaxf(v, acc[m][n][r]);
      // reduce across the 16 lanes of this half-wave (xor<16 stays in-half)
#pragma unroll
      for (int off = 1; off < 16; off <<= 1) v = fmaxf(v, __shfl_xor(v, off, 32));
      float s = 0.f;
#pragma unroll
      for (int n = 0; n < 4; ++n) s += __expf(acc[m][n][r] - v);
#pragma unroll
      for (int off = 1; off < 16; off <<= 1) s += __shfl_xor(s, off, 32);
      if (l15 == 0)  // lanes 0 and 16 write rows r and r+8
        mergeMS[nwave][mwave * 32 + m * 16 + r + half * 8] = make_float2(v, s);
    }
  }
  __syncthreads();

  if (t < TM) {  // merge the 2 n-wave columns, emit (m, s) partial for this vocab tile
    float2 a = mergeMS[0][t], b = mergeMS[1][t];
    float mm = fmaxf(a.x, b.x);
    float ss = a.y * __expf(a.x - mm) + b.y * __expf(b.x - mm);
    ws_ms[(size_t)(tbase + t) * VTILES + blockIdx.y] = make_float2(mm, ss);
  }
}

__global__ __launch_bounds__(256)
void ce_token_reduce(const float2* __restrict__ ws_ms, const float* __restrict__ ws_tgt,
                     float* __restrict__ partial)
{
  const int tok = blockIdx.x * 256 + threadIdx.x;
  float m = -3.402823466e38f, s = 0.f;
  const float2* p = ws_ms + (size_t)tok * VTILES;
  for (int v = 0; v < VTILES; ++v) {
    float2 q = p[v];
    float nm = fmaxf(m, q.x);
    s = s * __expf(m - nm) + q.y * __expf(q.x - nm);
    m = nm;
  }
  float loss = -(ws_tgt[tok] - m - __logf(s));

  __shared__ float red[256];
  red[threadIdx.x] = loss;
  __syncthreads();
  for (int off = 128; off > 0; off >>= 1) {
    if (threadIdx.x < off) red[threadIdx.x] += red[threadIdx.x + off];
    __syncthreads();
  }
  if (threadIdx.x == 0) partial[blockIdx.x] = red[0];
}

__global__ void ce_finalize(const float* __restrict__ partial, float* __restrict__ out)
{
  if (threadIdx.x == 0) {
    float s = 0.f;
    for (int i = 0; i < TOKENS / 256; ++i) s += partial[i];
    out[0] = s;
  }
}

extern "C" void kernel_launch(void* const* d_in, const int* in_sizes, int n_in,
                              void* d_out, int out_size, void* d_ws, size_t ws_size,
                              hipStream_t stream) {
  (void)in_sizes; (void)n_in; (void)out_size; (void)ws_size;
  const float* x   = (const float*)d_in[0];
  const float* w   = (const float*)d_in[1];
  const int*   tgt = (const int*)d_in[2];
  float*       out = (float*)d_out;

  char* ws = (char*)d_ws;
  float2* ws_ms   = (float2*)ws;                                                  // [TOKENS][VTILES]
  float*  ws_tgt  = (float*)(ws + (size_t)TOKENS * VTILES * sizeof(float2));      // [TOKENS]
  float*  partial = (float*)(ws + (size_t)TOKENS * VTILES * sizeof(float2)
                                + (size_t)TOKENS * sizeof(float));               // [TOKENS/256]

  ce_logits_partial<<<dim3(TTILES, VTILES), 256, 0, stream>>>(x, w, tgt, ws_ms, ws_tgt);
  ce_token_reduce<<<dim3(TOKENS / 256), 256, 0, stream>>>(ws_ms, ws_tgt, partial);
  ce_finalize<<<1, 32, 0, stream>>>(partial, out);
}
